// STAGEN_72258529788049
// MI455X (gfx1250) — matrix operand order repaired and checked by reading, not compiled, not run on hardware
//
#include <hip/hip_runtime.h>
#include <cstddef>

#define NE_  100000
#define NR_  256
#define H_   128
#define E_   250000
#define NTOT_ (NE_ + NR_)

typedef __attribute__((ext_vector_type(16))) __bf16 v16bf;
typedef __attribute__((ext_vector_type(8)))  float  v8f;

__device__ __forceinline__ unsigned short f2bf(float x) {
  unsigned u = __float_as_uint(x);
  unsigned r = u + 0x7FFFu + ((u >> 16) & 1u);
  return (unsigned short)(r >> 16);
}
__device__ __forceinline__ float sigm(float x) { return 1.0f / (1.0f + expf(-x)); }

// order-preserving float<->uint encoding for atomicMax-based segment max
__device__ __forceinline__ unsigned fenc(float f) {
  unsigned u = __float_as_uint(f);
  return (u & 0x80000000u) ? ~u : (u | 0x80000000u);
}
__device__ __forceinline__ float fdec(unsigned e) {
  unsigned u = (e & 0x80000000u) ? (e & 0x7FFFFFFFu) : ~e;
  return __uint_as_float(u);
}

// ---- WMMA fragment helpers (CDNA5 16x16x32 bf16, wave32) ----
// A-matrix 16x32 (16-bit): lanes 0-15: row=lane, K=kb..kb+7 (elts 0-7), kb+16..kb+23 (elts 8-15), kb=0
//                          lanes 16-31: row=lane-16, kb=8
__device__ __forceinline__ v16bf lds_afrag(const unsigned short* s, int lane, int k0, int stride) {
  int row = lane & 15;
  int kb  = (lane < 16) ? 0 : 8;
  const unsigned short* p = s + row * stride + k0 + kb;
  union { v16bf v; unsigned short u[16]; } f;
#pragma unroll
  for (int i = 0; i < 8; ++i) { f.u[i] = p[i]; f.u[8 + i] = p[16 + i]; }
  return f.v;
}
// B-matrix 32x16 (16-bit): lanes 0-15 hold K=0..7,16..23 (col=lane), lanes 16-31 hold K=8..15,24..31 (col=lane-16)
// Pre-packed so each lane reads 32 contiguous bytes.
__device__ __forceinline__ v16bf gbl_bfrag(const unsigned short* __restrict__ pk,
                                           int ntiles, int kt, int nt, int lane) {
  const unsigned short* p = pk + (((size_t)(kt * ntiles + nt) * 32 + lane) << 4);
  union { v16bf v; uint4 q[2]; } f;
  f.q[0] = *(const uint4*)(p);
  f.q[1] = *(const uint4*)(p + 8);
  return f.v;
}
__device__ __forceinline__ v8f wmma_bf(v16bf a, v16bf b, v8f c) {
  return __builtin_amdgcn_wmma_f32_16x16x32_bf16(false, a, false, b, (short)0, c, false, false);
}

// ---- pack a row-major f32 weight [K,N] into B-fragment bf16 blobs ----
__global__ void pack_b_kernel(const float* __restrict__ W, unsigned short* __restrict__ out,
                              int K, int N) {
  int i = blockIdx.x * blockDim.x + threadIdx.x;
  if (i >= K * N) return;
  int j = i & 15;
  int lane = (i >> 4) & 31;
  int t = i >> 9;
  int ntiles = N >> 4;
  int nt = t % ntiles;
  int kt = t / ntiles;
  int k = kt * 32 + ((j & 8) << 1) + ((lane < 16) ? 0 : 8) + (j & 7);
  int n = nt * 16 + (lane & 15);
  out[i] = f2bf(W[(size_t)k * N + n]);
}

// ---- init kernels ----
__global__ void init_seg_kernel(unsigned* __restrict__ smax, float* __restrict__ den) {
  int i = blockIdx.x * blockDim.x + threadIdx.x;
  if (i < NE_ * 8) { smax[i] = 0x007FFFFFu; /* fenc(-inf) */ den[i] = 0.0f; }
}
__global__ void init_h_kernel(float* __restrict__ h, const float* __restrict__ rel_emb) {
  size_t i = (size_t)blockIdx.x * blockDim.x + threadIdx.x;
  if (i >= (size_t)NTOT_ * H_) return;
  h[i] = (i < (size_t)NE_ * H_) ? 0.0f : rel_emb[i - (size_t)NE_ * H_];
}

// ---- Stage 1: per-edge DRGCN (16 edges per wave, 1 wave per block) ----
__global__ __launch_bounds__(32) void edge_kernel(
    const float* __restrict__ ent_emb, const float* __restrict__ drgcn_rel,
    const float* __restrict__ edge_time, const float* __restrict__ query_time,
    const int* __restrict__ srcp, const int* __restrict__ dstp, const int* __restrict__ etype,
    const float* __restrict__ time_coeff,
    const float* __restrict__ wg_W1, const float* __restrict__ wg_b1,
    const float* __restrict__ wg_W2, const float* __restrict__ wg_b2,
    const float* __restrict__ bq, const float* __restrict__ bk, const float* __restrict__ bv,
    const unsigned short* __restrict__ pkWg1a, const unsigned short* __restrict__ pkWg1b,
    const unsigned short* __restrict__ pkWq, const unsigned short* __restrict__ pkWk,
    const unsigned short* __restrict__ pkWv,
    float* __restrict__ scores, float* __restrict__ vbuf, unsigned* __restrict__ smax) {
  __shared__ unsigned short s_src[16 * 128];
  __shared__ unsigned short s_rel[16 * 128];
  __shared__ unsigned short s_dst[16 * 128];
  __shared__ unsigned short s_msg[16 * 128];
  __shared__ float s_srcf[16 * 128];
  __shared__ float s_relf[16 * 128];
  __shared__ float s_hid[16 * 128];
  __shared__ float s_k[16 * 128];
  __shared__ float s_q[16 * 128];
  __shared__ float s_tmk[16];
  __shared__ float s_dw[16];
  __shared__ int   s_dsti[16];

  int e0 = blockIdx.x * 16;
  int lane = threadIdx.x;
  float tc = fabsf(time_coeff[0]) + 1e-9f;

  if (lane < 16) {
    int e = e0 + lane;
    int d = dstp[e];
    s_dsti[lane] = d;
    s_tmk[lane] = sigm((query_time[d] - edge_time[e]) / tc);
  }
  __syncthreads();

  for (int idx = lane; idx < 2048; idx += 32) {
    int r = idx >> 7, c = idx & 127;
    int e = e0 + r;
    float sv = ent_emb[(size_t)srcp[e] * H_ + c];
    float rv = drgcn_rel[(size_t)etype[e] * H_ + c] * s_tmk[r];
    float dv = ent_emb[(size_t)s_dsti[r] * H_ + c];
    s_srcf[idx] = sv; s_relf[idx] = rv;
    s_src[idx] = f2bf(sv); s_rel[idx] = f2bf(rv); s_dst[idx] = f2bf(dv);
  }
  __syncthreads();

  // hidden = relu(src@W1a + rel@W1b + tmask*W1[256] + b1)
  int n = lane & 15, ro = (lane < 16) ? 0 : 8;
  for (int nt = 0; nt < 8; ++nt) {
    v8f acc = {};
#pragma unroll
    for (int kc = 0; kc < 4; ++kc) {
      acc = wmma_bf(lds_afrag(s_src, lane, kc * 32, 128), gbl_bfrag(pkWg1a, 8, kc, nt, lane), acc);
      acc = wmma_bf(lds_afrag(s_rel, lane, kc * 32, 128), gbl_bfrag(pkWg1b, 8, kc, nt, lane), acc);
    }
    int col = nt * 16 + n;
    float wl = wg_W1[256 * 128 + col], b1 = wg_b1[col];
#pragma unroll
    for (int r = 0; r < 8; ++r) {
      int row = r + ro;
      float v = acc[r] + b1 + s_tmk[row] * wl;
      s_hid[row * 128 + col] = fmaxf(v, 0.0f);
    }
  }
  __syncthreads();

  if (lane < 16) {
    float s = wg_b2[0];
    for (int c = 0; c < 128; ++c) s += s_hid[lane * 128 + c] * wg_W2[c];
    s_dw[lane] = sigm(s);
  }
  __syncthreads();

  for (int idx = lane; idx < 2048; idx += 32)
    s_msg[idx] = f2bf(s_srcf[idx] * s_relf[idx] * s_dw[idx >> 7]);
  __syncthreads();

  // q = dst@Wq+bq, k = msg@Wk+bk, v = msg@Wv+bv
  for (int nt = 0; nt < 8; ++nt) {
    v8f aq = {}, ak = {}, av = {};
#pragma unroll
    for (int kc = 0; kc < 4; ++kc) {
      v16bf ad = lds_afrag(s_dst, lane, kc * 32, 128);
      v16bf am = lds_afrag(s_msg, lane, kc * 32, 128);
      aq = wmma_bf(ad, gbl_bfrag(pkWq, 8, kc, nt, lane), aq);
      ak = wmma_bf(am, gbl_bfrag(pkWk, 8, kc, nt, lane), ak);
      av = wmma_bf(am, gbl_bfrag(pkWv, 8, kc, nt, lane), av);
    }
    int col = nt * 16 + n;
    float bql = bq[col], bkl = bk[col], bvl = bv[col];
#pragma unroll
    for (int r = 0; r < 8; ++r) {
      int row = r + ro;
      s_q[row * 128 + col] = aq[r] + bql;
      s_k[row * 128 + col] = ak[r] + bkl;
      vbuf[(size_t)(e0 + row) * 128 + col] = av[r] + bvl;
    }
  }
  __syncthreads();

  // per-head scores + segment max via atomics
  for (int u = lane; u < 128; u += 32) {
    int r = u >> 3, hd = u & 7;
    float s = 0.0f;
    for (int j = 0; j < 16; ++j)
      s += s_q[r * 128 + hd * 16 + j] * s_k[r * 128 + hd * 16 + j];
    s *= 0.25f;  // 1/sqrt(16)
    scores[(size_t)(e0 + r) * 8 + hd] = s;
    atomicMax(&smax[(size_t)s_dsti[r] * 8 + hd], fenc(s));
  }
}

// ---- Stage 2: exp + segment sum + weighted scatter of v ----
__global__ void pass2_kernel(const float* __restrict__ scores, const float* __restrict__ vbuf,
                             const unsigned* __restrict__ smax, float* __restrict__ den,
                             float* __restrict__ h, const int* __restrict__ dstp) {
  int t = blockIdx.x * blockDim.x + threadIdx.x;
  if (t >= E_ * 8) return;
  int e = t >> 3, hd = t & 7;
  int d = dstp[e];
  float ex = expf(scores[t] - fdec(smax[d * 8 + hd]));
  atomicAdd(&den[d * 8 + hd], ex);
  const float* vp = vbuf + (size_t)e * 128 + hd * 16;
  float* hp = h + (size_t)d * 128 + hd * 16;
#pragma unroll
  for (int j = 0; j < 16; ++j) atomicAdd(&hp[j], ex * vp[j]);
}

__global__ void pass3_kernel(float* __restrict__ h, const float* __restrict__ den) {
  size_t i = (size_t)blockIdx.x * blockDim.x + threadIdx.x;
  if (i >= (size_t)NE_ * H_) return;
  int node = (int)(i >> 7), hd = (int)((i >> 4) & 7);
  float dn = den[node * 8 + hd];
  h[i] = (dn > 0.0f) ? h[i] / dn : 0.0f;
}

// ---- Stage 3: ODE right-hand side f(h_in), h_in = h + sum c_i * k_i ----
__global__ __launch_bounds__(64) void f_eval_kernel(
    const float* __restrict__ h,
    const float* k1, const float* k2, const float* k3, const float* k4, const float* k5,
    float c1, float c2, float c3, float c4, float c5,
    float* __restrict__ kout,
    const unsigned short* __restrict__ pk_spW1, const unsigned short* __restrict__ pk_spW2,
    const unsigned short* __restrict__ pk_tmW1, const unsigned short* __restrict__ pk_tmW2,
    const float* __restrict__ sp_b1, const float* __restrict__ sp_b2,
    const float* __restrict__ tm_b1, const float* __restrict__ tm_b2,
    const float* __restrict__ fg_W, const float* __restrict__ fg_b) {
  __shared__ unsigned short s_h[2][16 * 128];
  __shared__ unsigned short s_sp1[2][16 * 256];
  __shared__ unsigned short s_tm1[2][16 * 64];
  __shared__ float s_sp[2][16 * 128];
  __shared__ float s_tm[2][16 * 128];
  __shared__ float s_g[2][2][16];

  int w = threadIdx.x >> 5, lane = threadIdx.x & 31;
  size_t base = ((size_t)blockIdx.x * 2 + w) * 16 * 128;
  int n = lane & 15, ro = (lane < 16) ? 0 : 8;

  for (int idx = lane; idx < 2048; idx += 32) {
    size_t gi = base + idx;
    float x = h[gi];
    if (k1) x += c1 * k1[gi];
    if (k2) x += c2 * k2[gi];
    if (k3) x += c3 * k3[gi];
    if (k4) x += c4 * k4[gi];
    if (k5) x += c5 * k5[gi];
    s_h[w][idx] = f2bf(x);
  }
  __syncthreads();

  // sp1 = gelu(h @ sp_W1 + b1)  [16,256]
  for (int nt = 0; nt < 16; ++nt) {
    v8f acc = {};
#pragma unroll
    for (int kc = 0; kc < 4; ++kc)
      acc = wmma_bf(lds_afrag(s_h[w], lane, kc * 32, 128), gbl_bfrag(pk_spW1, 16, kc, nt, lane), acc);
    int col = nt * 16 + n;
    float b = sp_b1[col];
#pragma unroll
    for (int r = 0; r < 8; ++r) {
      float x = acc[r] + b;
      float g = 0.5f * x * (1.0f + erff(x * 0.7071067811865475f));
      s_sp1[w][(r + ro) * 256 + col] = f2bf(g);
    }
  }
  __syncthreads();

  // sp = sp1 @ sp_W2 + b2  [16,128]
  for (int nt = 0; nt < 8; ++nt) {
    v8f acc = {};
#pragma unroll
    for (int kc = 0; kc < 8; ++kc)
      acc = wmma_bf(lds_afrag(s_sp1[w], lane, kc * 32, 256), gbl_bfrag(pk_spW2, 8, kc, nt, lane), acc);
    int col = nt * 16 + n;
    float b = sp_b2[col];
#pragma unroll
    for (int r = 0; r < 8; ++r) s_sp[w][(r + ro) * 128 + col] = acc[r] + b;
  }
  __syncthreads();

  // tm1 = softplus(h @ tm_W1 + b1)  [16,64]
  for (int nt = 0; nt < 4; ++nt) {
    v8f acc = {};
#pragma unroll
    for (int kc = 0; kc < 4; ++kc)
      acc = wmma_bf(lds_afrag(s_h[w], lane, kc * 32, 128), gbl_bfrag(pk_tmW1, 4, kc, nt, lane), acc);
    int col = nt * 16 + n;
    float b = tm_b1[col];
#pragma unroll
    for (int r = 0; r < 8; ++r) {
      float x = acc[r] + b;
      float sp = (x > 20.0f) ? x : log1pf(expf(x));
      s_tm1[w][(r + ro) * 64 + col] = f2bf(sp);
    }
  }
  __syncthreads();

  // tm = tm1 @ tm_W2 + b2  [16,128]
  for (int nt = 0; nt < 8; ++nt) {
    v8f acc = {};
#pragma unroll
    for (int kc = 0; kc < 2; ++kc)
      acc = wmma_bf(lds_afrag(s_tm1[w], lane, kc * 32, 64), gbl_bfrag(pk_tmW2, 8, kc, nt, lane), acc);
    int col = nt * 16 + n;
    float b = tm_b2[col];
#pragma unroll
    for (int r = 0; r < 8; ++r) s_tm[w][(r + ro) * 128 + col] = acc[r] + b;
  }
  __syncthreads();

  // g = sigmoid([sp, tm] @ fg_W + fg_b)  [16,2]
  if (lane < 16) {
    float g0 = fg_b[0], g1 = fg_b[1];
    for (int c = 0; c < 128; ++c) {
      float a = s_sp[w][lane * 128 + c], b = s_tm[w][lane * 128 + c];
      g0 += a * fg_W[c * 2 + 0] + b * fg_W[(128 + c) * 2 + 0];
      g1 += a * fg_W[c * 2 + 1] + b * fg_W[(128 + c) * 2 + 1];
    }
    s_g[w][0][lane] = sigm(g0);
    s_g[w][1][lane] = sigm(g1);
  }
  __syncthreads();

  for (int idx = lane; idx < 2048; idx += 32) {
    int r = idx >> 7;
    kout[base + idx] = s_g[w][0][r] * s_sp[w][idx] + s_g[w][1][r] * s_tm[w][idx];
  }
}

__global__ void combine_kernel(float* __restrict__ h,
                               const float* __restrict__ k1, const float* __restrict__ k3,
                               const float* __restrict__ k4, const float* __restrict__ k5,
                               const float* __restrict__ k6,
                               float b1, float b3, float b4, float b5, float b6) {
  size_t i = (size_t)blockIdx.x * blockDim.x + threadIdx.x;
  if (i >= (size_t)NTOT_ * H_) return;
  h[i] += b1 * k1[i] + b3 * k3[i] + b4 * k4[i] + b5 * k5[i] + b6 * k6[i];
}

extern "C" void kernel_launch(void* const* d_in, const int* in_sizes, int n_in,
                              void* d_out, int out_size, void* d_ws, size_t ws_size,
                              hipStream_t stream) {
  (void)in_sizes; (void)n_in; (void)out_size; (void)ws_size;
  const float* ent_emb    = (const float*)d_in[0];
  const float* rel_emb    = (const float*)d_in[1];
  const float* drgcn_rel  = (const float*)d_in[2];
  const float* Wq = (const float*)d_in[3];  const float* bq = (const float*)d_in[4];
  const float* Wk = (const float*)d_in[5];  const float* bk = (const float*)d_in[6];
  const float* Wv = (const float*)d_in[7];  const float* bv = (const float*)d_in[8];
  const float* time_coeff = (const float*)d_in[9];
  const float* wg_W1 = (const float*)d_in[10]; const float* wg_b1 = (const float*)d_in[11];
  const float* wg_W2 = (const float*)d_in[12]; const float* wg_b2 = (const float*)d_in[13];
  const float* sp_W1 = (const float*)d_in[14]; const float* sp_b1 = (const float*)d_in[15];
  const float* sp_W2 = (const float*)d_in[16]; const float* sp_b2 = (const float*)d_in[17];
  const float* tm_W1 = (const float*)d_in[18]; const float* tm_b1 = (const float*)d_in[19];
  const float* tm_W2 = (const float*)d_in[20]; const float* tm_b2 = (const float*)d_in[21];
  const float* fg_W  = (const float*)d_in[22]; const float* fg_b  = (const float*)d_in[23];
  const float* edge_time  = (const float*)d_in[24];
  const float* query_time = (const float*)d_in[25];
  const int* srcp  = (const int*)d_in[26];
  const int* dstp  = (const int*)d_in[27];
  const int* etype = (const int*)d_in[28];

  float* h = (float*)d_out;  // [NTOT, H] = [h_ent ; h_rel]

  // workspace layout
  char* ws = (char*)d_ws;
  size_t o = 0;
  float* vbuf = (float*)(ws + o);       o += (size_t)E_ * 128 * 4;
  float* scores = (float*)(ws + o);     o += (size_t)E_ * 8 * 4;
  unsigned* smax = (unsigned*)(ws + o); o += (size_t)NE_ * 8 * 4;
  float* den = (float*)(ws + o);        o += (size_t)NE_ * 8 * 4;
  float* K[6];
  for (int i = 0; i < 6; ++i) { K[i] = (float*)(ws + o); o += (size_t)NTOT_ * 128 * 4; }
  unsigned short* pk_spW1 = (unsigned short*)(ws + o); o += 128 * 256 * 2;
  unsigned short* pk_spW2 = (unsigned short*)(ws + o); o += 256 * 128 * 2;
  unsigned short* pk_tmW1 = (unsigned short*)(ws + o); o += 128 * 64 * 2;
  unsigned short* pk_tmW2 = (unsigned short*)(ws + o); o += 64 * 128 * 2;
  unsigned short* pkWq    = (unsigned short*)(ws + o); o += 128 * 128 * 2;
  unsigned short* pkWk    = (unsigned short*)(ws + o); o += 128 * 128 * 2;
  unsigned short* pkWv    = (unsigned short*)(ws + o); o += 128 * 128 * 2;
  unsigned short* pkWg1a  = (unsigned short*)(ws + o); o += 128 * 128 * 2;
  unsigned short* pkWg1b  = (unsigned short*)(ws + o); o += 128 * 128 * 2;

  auto PK = [&](const float* W, unsigned short* out, int Kd, int Nd) {
    pack_b_kernel<<<(Kd * Nd + 255) / 256, 256, 0, stream>>>(W, out, Kd, Nd);
  };
  PK(sp_W1, pk_spW1, 128, 256);
  PK(sp_W2, pk_spW2, 256, 128);
  PK(tm_W1, pk_tmW1, 128, 64);
  PK(tm_W2, pk_tmW2, 64, 128);
  PK(Wq, pkWq, 128, 128);
  PK(Wk, pkWk, 128, 128);
  PK(Wv, pkWv, 128, 128);
  PK(wg_W1, pkWg1a, 128, 128);
  PK(wg_W1 + 128 * 128, pkWg1b, 128, 128);

  init_seg_kernel<<<(NE_ * 8 + 255) / 256, 256, 0, stream>>>(smax, den);
  init_h_kernel<<<((size_t)NTOT_ * 128 + 255) / 256, 256, 0, stream>>>(h, rel_emb);

  edge_kernel<<<E_ / 16, 32, 0, stream>>>(
      ent_emb, drgcn_rel, edge_time, query_time, srcp, dstp, etype, time_coeff,
      wg_W1, wg_b1, wg_W2, wg_b2, bq, bk, bv,
      pkWg1a, pkWg1b, pkWq, pkWk, pkWv, scores, vbuf, smax);

  pass2_kernel<<<(E_ * 8 + 255) / 256, 256, 0, stream>>>(scores, vbuf, smax, den, h, dstp);
  pass3_kernel<<<(NE_ * 128 + 255) / 256, 256, 0, stream>>>(h, den);

  // dopri5 fixed-step, dt = 1/8
  const float DT = 0.125f;
  const float a21 = 0.2f;
  const float a31 = 3.f / 40, a32 = 9.f / 40;
  const float a41 = 44.f / 45, a42 = -56.f / 15, a43 = 32.f / 9;
  const float a51 = 19372.f / 6561, a52 = -25360.f / 2187, a53 = 64448.f / 6561, a54 = -212.f / 729;
  const float a61 = 9017.f / 3168, a62 = -355.f / 33, a63 = 46732.f / 5247, a64 = 49.f / 176,
              a65 = -5103.f / 18656;
  const float b1 = 35.f / 384, b3 = 500.f / 1113, b4 = 125.f / 192, b5 = -2187.f / 6784,
              b6 = 11.f / 84;

  auto FE = [&](const float* q1, const float* q2, const float* q3, const float* q4,
                const float* q5, float c1, float c2, float c3, float c4, float c5,
                float* kout) {
    f_eval_kernel<<<NTOT_ / 32, 64, 0, stream>>>(
        h, q1, q2, q3, q4, q5, c1, c2, c3, c4, c5, kout,
        pk_spW1, pk_spW2, pk_tmW1, pk_tmW2, sp_b1, sp_b2, tm_b1, tm_b2, fg_W, fg_b);
  };

  for (int step = 0; step < 8; ++step) {
    FE(nullptr, nullptr, nullptr, nullptr, nullptr, 0, 0, 0, 0, 0, K[0]);
    FE(K[0], nullptr, nullptr, nullptr, nullptr, DT * a21, 0, 0, 0, 0, K[1]);
    FE(K[0], K[1], nullptr, nullptr, nullptr, DT * a31, DT * a32, 0, 0, 0, K[2]);
    FE(K[0], K[1], K[2], nullptr, nullptr, DT * a41, DT * a42, DT * a43, 0, 0, K[3]);
    FE(K[0], K[1], K[2], K[3], nullptr, DT * a51, DT * a52, DT * a53, DT * a54, 0, K[4]);
    FE(K[0], K[1], K[2], K[3], K[4], DT * a61, DT * a62, DT * a63, DT * a64, DT * a65, K[5]);
    combine_kernel<<<((size_t)NTOT_ * 128 + 255) / 256, 256, 0, stream>>>(
        h, K[0], K[2], K[3], K[4], K[5], DT * b1, DT * b3, DT * b4, DT * b5, DT * b6);
  }
}
